// DMlp_46823733461564
// MI455X (gfx1250) — matrix operand order repaired and checked by reading, not compiled, not run on hardware
//
#include <hip/hip_runtime.h>
#include <hip/hip_bf16.h>

// ---------------------------------------------------------------------------
// DMlp fused pipeline for MI455X (gfx1250, wave32, WMMA + TDM async tiles)
//   Kernel 0a: w1 (f32, k=c*25+p) -> w1p (bf16, k'=p*64+c)        [one-time]
//   Kernel 0b: w2 (f32) -> w2bf (bf16)                             [one-time]
//   Kernel 1 : gather fused into A staging (one mask position per K-chunk),
//              B tile via TENSOR_LOAD_TO_LDS, double-buffered, 1 barrier/step
//              GEMM1 (bf16 WMMA, f32 acc) -> bias + exact GELU -> h (bf16)
//   Kernel 2 : A and B tiles both via TENSOR_LOAD_TO_LDS, double-buffered
//              GEMM2 (bf16 WMMA, f32 acc) -> bias -> out (f32)
// ---------------------------------------------------------------------------

typedef __attribute__((ext_vector_type(8)))  __bf16 v8bf;
typedef __attribute__((ext_vector_type(16))) __bf16 v16bf;
typedef __attribute__((ext_vector_type(8)))  float  v8f;
typedef unsigned int u32x4 __attribute__((ext_vector_type(4)));
typedef int          i32x4 __attribute__((ext_vector_type(4)));
typedef int          i32x8 __attribute__((ext_vector_type(8)));

union BF16x16 { v16bf v; v8bf h[2]; __bf16 e[16]; };

#define NPIX  16384     // 128*128
#define LIN   576       // input channel dim of x
#define K1    1600      // 25 * 64, p-major k' ordering
#define N1    1024      // hidden
#define K2    1024
#define N2    576
#define MTOT  65536     // B * NPIX

// 25 masked (di,dj) positions of the 7x7 mask, packed di*16+dj (row-major).
__constant__ int c_dij[25] = {
    0x00, 0x03, 0x06,
    0x11, 0x13, 0x15,
    0x22, 0x23, 0x24,
    0x30, 0x31, 0x32, 0x33, 0x34, 0x35, 0x36,
    0x42, 0x43, 0x44,
    0x51, 0x53, 0x55,
    0x60, 0x63, 0x66
};

__device__ __forceinline__ int reflect384(int v) {
    return v < 0 ? -v : (v > 383 ? 766 - v : v);
}

// ---------------------------------------------------------------------------
// TDM: async 2D tile (tile_w x tile_h bf16 elements, row stride in elements)
// from global memory into LDS (row-major, row pitch = tile_w). D# per ISA
// 8.3/8.4: group0 = {count,lds_addr,global_addr,type=2},
// group1 = {data_size=2B, tensor dims, tile dims, dim0 stride}. 2D tensor ->
// groups 2/3 (and the trailing group) zero. Tracked on TENSORcnt.
// ---------------------------------------------------------------------------
__device__ __forceinline__ void tdm_load_2d(const void* gptr, unsigned lds_off,
                                            int tile_w, int tile_h, int row_stride)
{
    const unsigned long long ga = (unsigned long long)(uintptr_t)gptr;
    u32x4 g0;
    g0[0] = 1u;                                               // count=1
    g0[1] = lds_off;                                          // LDS byte address
    g0[2] = (unsigned)(ga & 0xFFFFFFFFu);                     // global_addr[31:0]
    g0[3] = (unsigned)((ga >> 32) & 0x1FFFFFFu) | (2u << 30); // [56:32] | type=2
    i32x8 g1;
    g1[0] = (1 << 16);                                        // data_size=1 -> 2B
    g1[1] = (tile_w & 0xFFFF) << 16;                          // tensor_dim0[15:0]
    g1[2] = ((unsigned)tile_h) << 16;                         // dim0 hi=0 | tensor_dim1 lo
    g1[3] = ((unsigned)tile_w) << 16;                         // dim1 hi=0 | tile_dim0
    g1[4] = (tile_h & 0xFFFF);                                // tile_dim1 | tile_dim2=0
    g1[5] = row_stride;                                       // tensor_dim0_stride[31:0]
    g1[6] = 0;
    g1[7] = 0;
    const i32x4 z4 = {0, 0, 0, 0};
    const i32x8 z8 = {0, 0, 0, 0, 0, 0, 0, 0};
    __builtin_amdgcn_tensor_load_to_lds(g0, g1, z4, z4, z8, 0);
}

// ---------------------------------------------------------------------------
// Kernel 0a: permute+convert w1: w1p[o*1600 + p*64 + c] = bf16(w1[o*1600 + c*25 + p])
// ---------------------------------------------------------------------------
__global__ __launch_bounds__(256)
void permute_w1(const float* __restrict__ w1, __bf16* __restrict__ w1p)
{
    const int i   = blockIdx.x * 256 + threadIdx.x;
    const int o   = i / K1;
    const int rem = i - o * K1;
    const int p   = rem >> 6;
    const int c   = rem & 63;
    w1p[i] = (__bf16)w1[o * K1 + c * 25 + p];
}

__global__ __launch_bounds__(256)
void convert_w2(const float* __restrict__ w2, __bf16* __restrict__ w2bf)
{
    const int i = blockIdx.x * 256 + threadIdx.x;
    w2bf[i] = (__bf16)w2[i];
}

// ---------------------------------------------------------------------------
// Kernel 1: fused gather + GEMM1 + GELU.  Grid: (MTOT/128, N1/128), 256 thr.
// Double-buffered: A staged by gather VALU, B by TDM. One barrier per K-step.
// ---------------------------------------------------------------------------
__global__ __launch_bounds__(256)
void gemm1_gather_gelu(const float* __restrict__ x,
                       const __bf16* __restrict__ w1p,
                       const float* __restrict__ b1,
                       __bf16* __restrict__ hbuf)
{
    __shared__ __align__(32) __bf16 Alds[2][128 * 32];  // 2 x 8 KB
    __shared__ __align__(32) __bf16 Blds[2][128 * 32];  // 2 x 8 KB

    const int tid  = threadIdx.x;
    const int wave = tid >> 5;
    const int lane = tid & 31;
    const int m0   = blockIdx.x * 128;
    const int nb   = blockIdx.y * 128;

    v8f acc[8] = {};

    // gather assignment: 2 threads per row, 16 elements each
    const int a_row = tid >> 1;
    const int a_kc  = (tid & 1) * 16;
    const int mrow  = m0 + a_row;
    const int bimg  = mrow >> 14;
    const int n     = mrow & (NPIX - 1);
    const int ii    = n >> 7;
    const int jj    = n & 127;
    const int xbase = bimg * NPIX * LIN;

    const __bf16* wblock = &w1p[nb * K1];

    // fragment addressing
    const int mloc = wave * 16 + (lane & 15);
    const int koff = (lane >> 4) * 16;      // A: lanes 16-31 hold K=16..31
    const int kk0  = (lane >> 4) * 16;      // B: lanes 16-31 hold K=16..31

    auto geom = [&](int p) -> const float* {
        const int dd = c_dij[p];
        const int ry = reflect384(3 * ii + (dd >> 4) - 2);
        const int rx = reflect384(3 * jj + (dd & 15) - 2);
        const int hh = ry / 3, ri = ry - 3 * hh;
        const int wq = rx / 3, rj = rx - 3 * wq;
        return &x[xbase + ((hh << 7) + wq) * LIN + ri * 3 + rj];
    };
    auto stageA = [&](const float* base, int halfc, int buf) {
        const float* apc = base + (halfc * 32 + a_kc) * 9;
        BF16x16 pack;
        #pragma unroll
        for (int kk = 0; kk < 16; ++kk)
            pack.e[kk] = (__bf16)apc[kk * 9];
        *(v16bf*)&Alds[buf][a_row * 32 + a_kc] = pack.v;
    };

    // prologue: stage step 0 into buffer 0
    const float* ap = geom(0);
    stageA(ap, 0, 0);
    if (tid < 32)
        tdm_load_2d(wblock, (unsigned)(uintptr_t)&Blds[0][0], 32, 128, K1);

    for (int it = 0; it < 50; ++it) {        // k' = it*32, p = it>>1
        const int buf = it & 1;
        __builtin_amdgcn_s_wait_tensorcnt(0);   // B(it) landed (wave0; no-op others)
        __syncthreads();                        // A(it) visible; buf^1 free
        const int nxt = it + 1;
        if (nxt < 50) {
            if ((nxt & 1) == 0) ap = geom(nxt >> 1);   // new mask position
            stageA(ap, nxt & 1, buf ^ 1);
            if (tid < 32)
                tdm_load_2d(wblock + nxt * 32,
                            (unsigned)(uintptr_t)&Blds[buf ^ 1][0], 32, 128, K1);
        }
        // compute: 8 WMMA per wave per K-step (overlaps TDM + gather loads)
        BF16x16 afrag;
        afrag.h[0] = *(const v8bf*)&Alds[buf][mloc * 32 + koff];
        afrag.h[1] = *(const v8bf*)&Alds[buf][mloc * 32 + 8 + koff];
        #pragma unroll
        for (int nt = 0; nt < 8; ++nt) {
            v16bf bfrag = *(const v16bf*)&Blds[buf][(nt * 16 + (lane & 15)) * 32 + kk0];
            acc[nt] = __builtin_amdgcn_wmma_f32_16x16x32_bf16(
                false, afrag.v, false, bfrag, (short)0, acc[nt], false, false);
        }
    }

    // epilogue: bias + exact GELU -> bf16 h
    const int lrow = wave * 16 + (lane >> 4) * 8;
    const int lcol = lane & 15;
    #pragma unroll
    for (int nt = 0; nt < 8; ++nt) {
        const int col  = nb + nt * 16 + lcol;
        const float bs = b1[col];
        #pragma unroll
        for (int v = 0; v < 8; ++v) {
            const float z = acc[nt][v] + bs;
            const float g = 0.5f * z * (1.0f + erff(z * 0.70710678118654752f));
            hbuf[(m0 + lrow + v) * N1 + col] = (__bf16)g;
        }
    }
}

// ---------------------------------------------------------------------------
// Kernel 2: GEMM2 + bias.  Grid: (MTOT/128, N2/96), 256 threads.
// Both tiles via TDM, double-buffered; main loop is ds_load + wmma only.
// ---------------------------------------------------------------------------
__global__ __launch_bounds__(256)
void gemm2_bias(const __bf16* __restrict__ hbuf,
                const __bf16* __restrict__ w2bf,
                const float* __restrict__ b2,
                float* __restrict__ out)
{
    __shared__ __align__(32) __bf16 Alds[2][128 * 32];  // 2 x 8 KB
    __shared__ __align__(32) __bf16 Blds[2][96 * 32];   // 2 x 6 KB

    const int tid  = threadIdx.x;
    const int wave = tid >> 5;
    const int lane = tid & 31;
    const int m0   = blockIdx.x * 128;
    const int nb   = blockIdx.y * 96;

    v8f acc[6] = {};

    const __bf16* hblock = &hbuf[m0 * K2];
    const __bf16* wblock = &w2bf[nb * K2];

    const int mloc = wave * 16 + (lane & 15);
    const int koff = (lane >> 4) * 16;
    const int kk0  = (lane >> 4) * 16;

    if (tid < 32) {
        tdm_load_2d(hblock, (unsigned)(uintptr_t)&Alds[0][0], 32, 128, K2);
        tdm_load_2d(wblock, (unsigned)(uintptr_t)&Blds[0][0], 32, 96,  K2);
    }

    for (int it = 0; it < K2 / 32; ++it) {
        const int buf = it & 1;
        __builtin_amdgcn_s_wait_tensorcnt(0);
        __syncthreads();
        if (it + 1 < K2 / 32 && tid < 32) {
            tdm_load_2d(hblock + (it + 1) * 32,
                        (unsigned)(uintptr_t)&Alds[buf ^ 1][0], 32, 128, K2);
            tdm_load_2d(wblock + (it + 1) * 32,
                        (unsigned)(uintptr_t)&Blds[buf ^ 1][0], 32, 96,  K2);
        }
        BF16x16 afrag;
        afrag.h[0] = *(const v8bf*)&Alds[buf][mloc * 32 + koff];
        afrag.h[1] = *(const v8bf*)&Alds[buf][mloc * 32 + 8 + koff];
        #pragma unroll
        for (int nt = 0; nt < 6; ++nt) {
            v16bf bfrag = *(const v16bf*)&Blds[buf][(nt * 16 + (lane & 15)) * 32 + kk0];
            acc[nt] = __builtin_amdgcn_wmma_f32_16x16x32_bf16(
                false, afrag.v, false, bfrag, (short)0, acc[nt], false, false);
        }
    }

    const int lrow = wave * 16 + (lane >> 4) * 8;
    const int lcol = lane & 15;
    #pragma unroll
    for (int nt = 0; nt < 6; ++nt) {
        const int col  = nb + nt * 16 + lcol;
        const float bs = b2[col];
        #pragma unroll
        for (int v = 0; v < 8; ++v)
            out[(m0 + lrow + v) * N2 + col] = acc[nt][v] + bs;
    }
}

// ---------------------------------------------------------------------------
extern "C" void kernel_launch(void* const* d_in, const int* in_sizes, int n_in,
                              void* d_out, int out_size, void* d_ws, size_t ws_size,
                              hipStream_t stream) {
    const float* x  = (const float*)d_in[0];
    const float* w1 = (const float*)d_in[1];
    const float* b1 = (const float*)d_in[2];
    const float* w2 = (const float*)d_in[3];
    const float* b2 = (const float*)d_in[4];
    float* out = (float*)d_out;

    __bf16* hbuf = (__bf16*)d_ws;                       // 134.2 MB
    __bf16* w1p  = hbuf + (size_t)MTOT * N1;            // 3.28 MB
    __bf16* w2bf = w1p  + (size_t)N1 * K1;              // 1.18 MB

    dim3 blk(256);
    permute_w1<<<dim3((N1 * K1) / 256), blk, 0, stream>>>(w1, w1p);
    convert_w2<<<dim3((N2 * K2) / 256), blk, 0, stream>>>(w2, w2bf);

    dim3 g1(MTOT / 128, N1 / 128);  // 512 x 8
    dim3 g2(MTOT / 128, N2 / 96);   // 512 x 6
    gemm1_gather_gelu<<<g1, blk, 0, stream>>>(x, w1p, b1, hbuf);
    gemm2_bias<<<g2, blk, 0, stream>>>(hbuf, w2bf, b2, out);
}
